// TorchRQ_36215164240446
// MI455X (gfx1250) — matrix operand order repaired and checked by reading, not compiled, not run on hardware
//
#include <hip/hip_runtime.h>
#include <hip/hip_bf16.h>

// ---------------- problem constants ----------------
#define B_ROWS   65536
#define DIM      512
#define M_STAGES 8
#define K_CODES  1024

// ---------------- vector types ----------------
typedef __attribute__((ext_vector_type(16))) __bf16         v16bf;
typedef __attribute__((ext_vector_type(16))) unsigned short v16us;
typedef __attribute__((ext_vector_type(8)))  float          v8f;
typedef __attribute__((ext_vector_type(4))) unsigned int    uint32x4;
typedef __attribute__((ext_vector_type(8)))  int            int32x8;
typedef __attribute__((ext_vector_type(4)))  int            int32x4;

// ---------------- LDS layout (dynamic) ----------------
#define RESID_STRIDE 516                         // floats/row (512 + 4 pad -> conflict-free A reads)
#define RESID_BYTES  (128 * RESID_STRIDE * 4)    // 264192
#define CB_ROW_US    528                         // ushorts/row (512 + 16 pad, matches TDM pad)
#define CB_TILE_US   (16 * CB_ROW_US)            // one staged 16-code tile
#define CB_BYTES     (2 * CB_TILE_US * 2)        // double buffered: 33792
#define CBSQ_BYTES   (K_CODES * 4)               // 4096
#define CODES_BYTES  (8 * 16 * 4)                // 512
#define LDS_TOTAL    (RESID_BYTES + CB_BYTES + CBSQ_BYTES + CODES_BYTES)

__device__ __forceinline__ unsigned short f2bf(float f) {
    unsigned int u = __builtin_bit_cast(unsigned int, f);
    unsigned int r = u + 0x7FFFu + ((u >> 16) & 1u);   // round-to-nearest-even
    return (unsigned short)(r >> 16);
}

// ---------------- TDM: DMA one 16x512 bf16 codebook tile -> LDS ----------------
// D# per CDNA5 ISA. 2D tile, data_size=2B,
// pad_enable: 32B pad after every 1024B row -> LDS row stride 1056B = CB_ROW_US ushorts.
__device__ __forceinline__ void tdm_issue_tile(unsigned long long gaddr, unsigned lds_off_bytes) {
    uint32x4 g0;
    g0[0] = 1u;                                            // count=1, user descriptor
    g0[1] = lds_off_bytes;                                 // lds_addr
    g0[2] = (unsigned)(gaddr & 0xFFFFFFFFull);             // global_addr[31:0]
    g0[3] = (unsigned)((gaddr >> 32) & 0x1FFFFFFull) | (2u << 30);  // addr[56:32] | type=2
    int32x8 g1;
    g1[0] = (int)((1u << 16) | (1u << 20) | (7u << 22) | (7u << 25)); // data_size=2B, pad_en, interval=256dw, amount=8dw
    g1[1] = (int)(512u << 16);     // tensor_dim0 = 512 (low16 in [31:16])
    g1[2] = (int)(16u  << 16);     // tensor_dim0 hi=0 | tensor_dim1 = 16
    g1[3] = (int)(512u << 16);     // tensor_dim1 hi=0 | tile_dim0 = 512
    g1[4] = 16;                    // tile_dim1 = 16 | tile_dim2 = 0
    g1[5] = 512;                   // tensor_dim0_stride = 512 elements
    g1[6] = 0;
    g1[7] = 0;
    int32x4 gz4 = {0, 0, 0, 0};                 // groups 2/3 unused (2D tile)
    int32x8 gz8 = {0, 0, 0, 0, 0, 0, 0, 0};     // trailing descriptor words (clang-23 6-arg form)
    __builtin_amdgcn_tensor_load_to_lds(g0, g1, gz4, gz4, gz8, 0);
}

// ---------------- kernel 0: zero the MSE accumulator ----------------
__global__ void rq_zero(float* mse) {
    if (threadIdx.x == 0 && blockIdx.x == 0) *mse = 0.0f;
}

// ---------------- kernel 1: codebooks f32 -> bf16 + per-code squared norms ----------------
__global__ void rq_prep(const float* __restrict__ cb,
                        unsigned short* __restrict__ cb_bf,
                        float* __restrict__ cbsq) {
    const int row = blockIdx.x;                       // 0 .. M*K-1
    const float* src = cb + (size_t)row * DIM;
    unsigned short* dst = cb_bf + (size_t)row * DIM;
    const int t = threadIdx.x;
    float a = src[t];
    float b = src[t + 256];
    dst[t]       = f2bf(a);
    dst[t + 256] = f2bf(b);
    float p = a * a + b * b;
    for (int off = 16; off; off >>= 1) p += __shfl_xor(p, off, 32);
    __shared__ float wsum[8];
    if ((t & 31) == 0) wsum[t >> 5] = p;
    __syncthreads();
    if (t == 0) {
        float s = 0.f;
        for (int i = 0; i < 8; ++i) s += wsum[i];
        cbsq[row] = s;
    }
}

// ---------------- kernel 2: fused RQ encode + decode + loss ----------------
extern __shared__ char smem_raw[];

__launch_bounds__(256, 1)
__global__ void rq_main(const float* __restrict__ x,
                        const float* __restrict__ cb_f32,
                        const unsigned short* __restrict__ cb_bf,
                        const float* __restrict__ cbsq,
                        int* __restrict__ codes_out,
                        float* __restrict__ xhat_out,
                        float* __restrict__ mse_out) {
    float*          resid    = (float*)smem_raw;
    unsigned short* cb_sm    = (unsigned short*)(smem_raw + RESID_BYTES);
    float*          cbsq_sm  = (float*)(smem_raw + RESID_BYTES + CB_BYTES);
    int*            codes_sm = (int*)(smem_raw + RESID_BYTES + CB_BYTES + CBSQ_BYTES);

    const int tid  = threadIdx.x;
    const int lane = tid & 31;
    const int wv   = tid >> 5;           // 8 waves, each owns 16 rows
    const int mrow = lane & 15;          // A-matrix row / B-matrix column within a 16x16 tile
    const int kh   = lane >> 4;          // which K-half this lane holds
    const int row0 = blockIdx.x * 128;   // global base row of this workgroup

    // ---- load x tile into residual (xq starts at 0 so resid == x) ----
    for (int i = tid; i < 128 * (DIM / 4); i += 256) {
        int r = i / (DIM / 4);
        int c = (i % (DIM / 4)) * 4;
        float4 v = *(const float4*)(x + (size_t)(row0 + r) * DIM + c);
        *(float4*)(resid + r * RESID_STRIDE + c) = v;
    }
    __syncthreads();

    for (int m = 0; m < M_STAGES; ++m) {
        // stage the per-code squared norms
        for (int i = tid; i < K_CODES; i += 256)
            cbsq_sm[i] = cbsq[m * K_CODES + i];

        // TDM prologue: issue tile 0 -> buffer 0 (wave 0 drives the DMA)
        const unsigned long long cb_stage =
            (unsigned long long)(size_t)(cb_bf + (size_t)m * K_CODES * DIM);
        if (wv == 0) tdm_issue_tile(cb_stage, RESID_BYTES);

        // ---- build A fragments (16 rows x 512) as bf16, held in VGPRs all stage ----
        v16bf afr[16];
        {
            const float* rp = resid + (wv * 16 + mrow) * RESID_STRIDE;
            #pragma unroll
            for (int ks = 0; ks < 16; ++ks) {
                const int d0 = ks * 32 + kh * 16;     // lane holds K = d0..d0+15
                float4 p0 = *(const float4*)(rp + d0);
                float4 p1 = *(const float4*)(rp + d0 + 4);
                float4 p2 = *(const float4*)(rp + d0 + 8);
                float4 p3 = *(const float4*)(rp + d0 + 12);
                v16us u;
                u[0]  = f2bf(p0.x); u[1]  = f2bf(p0.y); u[2]  = f2bf(p0.z); u[3]  = f2bf(p0.w);
                u[4]  = f2bf(p1.x); u[5]  = f2bf(p1.y); u[6]  = f2bf(p1.z); u[7]  = f2bf(p1.w);
                u[8]  = f2bf(p2.x); u[9]  = f2bf(p2.y); u[10] = f2bf(p2.z); u[11] = f2bf(p2.w);
                u[12] = f2bf(p3.x); u[13] = f2bf(p3.y); u[14] = f2bf(p3.z); u[15] = f2bf(p3.w);
                afr[ks] = __builtin_bit_cast(v16bf, u);
            }
        }

        float bv[8];
        int   bi[8];
        #pragma unroll
        for (int v = 0; v < 8; ++v) { bv[v] = 3.0e38f; bi[v] = 0; }

        // ---- sweep all 1024 codes in 16-code tiles, double-buffered TDM ----
        for (int nt = 0; nt < 64; ++nt) {
            if (wv == 0) {
                if (nt + 1 < 64) {
                    tdm_issue_tile(cb_stage + (size_t)(nt + 1) * 16 * DIM * 2,
                                   RESID_BYTES + ((nt + 1) & 1) * CB_TILE_US * 2);
                    __builtin_amdgcn_s_wait_tensorcnt(1);   // oldest (tile nt) complete
                } else {
                    __builtin_amdgcn_s_wait_tensorcnt(0);
                }
            }
            __syncthreads();   // publish staged tile to all waves

            const unsigned short* bp =
                cb_sm + (nt & 1) * CB_TILE_US + mrow * CB_ROW_US + kh * 16;

            // fetch the distance-bias term early so its wait doesn't trail the WMMAs
            const int   code = nt * 16 + mrow;
            const float cq   = cbsq_sm[code];

            // batch-load the whole B tile (32 ds_load_b128) so one wait covers all,
            // then issue the 16 WMMAs back-to-back on the accumulator chain
            v16us bu[16];
            #pragma unroll
            for (int ks = 0; ks < 16; ++ks)
                bu[ks] = *(const v16us*)(bp + ks * 32);

            v8f acc = {0.f, 0.f, 0.f, 0.f, 0.f, 0.f, 0.f, 0.f};
            #pragma unroll
            for (int ks = 0; ks < 16; ++ks) {
                v16bf bfr = __builtin_bit_cast(v16bf, bu[ks]);
                acc = __builtin_amdgcn_wmma_f32_16x16x32_bf16(
                          false, afr[ks], false, bfr, (short)0, acc, false, false);
            }

            #pragma unroll
            for (int v = 0; v < 8; ++v) {
                float cand = cq - 2.0f * acc[v];    // ||r-c||^2 up to const
                if (cand < bv[v]) { bv[v] = cand; bi[v] = code; }  // codes increase -> first-min kept
            }
            __syncthreads();   // everyone done reading this buffer before it is overwritten
        }

        // ---- argmin across the 16 code-columns held by lanes of each half-wave ----
        #pragma unroll
        for (int v = 0; v < 8; ++v) {
            #pragma unroll
            for (int off = 1; off < 16; off <<= 1) {
                float ov = __shfl_xor(bv[v], off, 32);
                int   oi = __shfl_xor(bi[v], off, 32);
                if (ov < bv[v] || (ov == bv[v] && oi < bi[v])) { bv[v] = ov; bi[v] = oi; }
            }
        }
        if (mrow == 0) {   // lanes 0 and 16: rows v and v+8 of this wave's tile
            #pragma unroll
            for (int v = 0; v < 8; ++v) {
                int r = v + 8 * kh;
                codes_sm[wv * 16 + r] = bi[v];
                codes_out[(size_t)m * B_ROWS + row0 + wv * 16 + r] = bi[v];
            }
        }
        __syncthreads();

        // ---- gather chosen codeword (f32, L2-resident) and update residual ----
        for (int r = 0; r < 16; ++r) {
            int c = codes_sm[wv * 16 + r];
            const float4* src = (const float4*)(cb_f32 + ((size_t)m * K_CODES + c) * DIM);
            float4* dst = (float4*)(resid + (wv * 16 + r) * RESID_STRIDE);
            #pragma unroll
            for (int j = 0; j < 4; ++j) {
                int idx = lane + j * 32;
                float4 cv = src[idx];
                float4 rv = dst[idx];
                rv.x -= cv.x; rv.y -= cv.y; rv.z -= cv.z; rv.w -= cv.w;
                dst[idx] = rv;
            }
        }
        __syncthreads();
    }

    // ---- outputs: x_hat = x - resid ; mse = mean(resid^2) ----
    float part = 0.f;
    for (int i = tid; i < 128 * (DIM / 4); i += 256) {
        int r = i / (DIM / 4);
        int c = (i % (DIM / 4)) * 4;
        float4 xv = *(const float4*)(x + (size_t)(row0 + r) * DIM + c);
        float4 rv = *(const float4*)(resid + r * RESID_STRIDE + c);
        float4 h;
        h.x = xv.x - rv.x; h.y = xv.y - rv.y; h.z = xv.z - rv.z; h.w = xv.w - rv.w;
        *(float4*)(xhat_out + (size_t)(row0 + r) * DIM + c) = h;
        part += rv.x * rv.x + rv.y * rv.y + rv.z * rv.z + rv.w * rv.w;
    }
    for (int off = 16; off; off >>= 1) part += __shfl_xor(part, off, 32);
    if (lane == 0) cbsq_sm[wv] = part;   // cbsq_sm no longer needed: reuse as scratch
    __syncthreads();
    if (tid == 0) {
        float s = 0.f;
        for (int i = 0; i < 8; ++i) s += cbsq_sm[i];
        atomicAdd(mse_out, s * (1.0f / (65536.0f * 512.0f)));
    }
}

// ---------------- host glue ----------------
extern "C" void kernel_launch(void* const* d_in, const int* in_sizes, int n_in,
                              void* d_out, int out_size, void* d_ws, size_t ws_size,
                              hipStream_t stream) {
    (void)in_sizes; (void)n_in; (void)out_size; (void)ws_size;
    const float* x  = (const float*)d_in[0];                 // [B, D]
    const float* cb = (const float*)d_in[1];                 // [M, K, D]

    unsigned short* cb_bf = (unsigned short*)d_ws;           // 8 MB bf16 codebooks
    float* cbsq = (float*)((char*)d_ws + (size_t)M_STAGES * K_CODES * DIM * 2);

    int*   codes_out = (int*)d_out;                                    // [M, B] int32
    float* xhat_out  = (float*)d_out + (size_t)M_STAGES * B_ROWS;      // [B, D]
    float* mse_out   = xhat_out + (size_t)B_ROWS * DIM;                // scalar

    rq_zero<<<1, 64, 0, stream>>>(mse_out);
    rq_prep<<<M_STAGES * K_CODES, 256, 0, stream>>>(cb, cb_bf, cbsq);
    rq_main<<<B_ROWS / 128, 256, LDS_TOTAL, stream>>>(
        x, cb, cb_bf, cbsq, codes_out, xhat_out, mse_out);
}
